// LlamaAttention_53532472377667
// MI455X (gfx1250) — matrix operand order repaired and checked by reading, not compiled, not run on hardware
//
#include <hip/hip_runtime.h>
#include <hip/hip_bf16.h>

// ---------------------------------------------------------------------------
// LlamaAttention forward for gfx1250 (MI455X): bf16 WMMA everywhere, f32 accum.
// Pipeline: f32->bf16 converts, LDS-staged (async-to-LDS, double-buffered)
// QKV GEMMs, RoPE+relayout, 1-wave flash attention, output GEMM.
// Workspace usage: 208 MB.
// ---------------------------------------------------------------------------

typedef __bf16 bf16;
typedef bf16  bf16x16 __attribute__((ext_vector_type(16)));
typedef bf16  bf16x8  __attribute__((ext_vector_type(8)));
typedef float f32x8   __attribute__((ext_vector_type(8)));
typedef int   i32x4   __attribute__((ext_vector_type(4)));

#define SEQ      4096
#define HIDDEN   4096
#define NHEADS   32
#define NKV      8
#define HDIM     128
#define NREP     (NHEADS / NKV)

#if defined(__HIP_DEVICE_COMPILE__) && \
    __has_builtin(__builtin_amdgcn_global_load_async_to_lds_b128) && \
    __has_builtin(__builtin_amdgcn_s_wait_asynccnt)
#define HAS_ASYNC_LDS 1
#else
#define HAS_ASYNC_LDS 0
#endif

__device__ __forceinline__ f32x8 wmma_bf16(bf16x16 a, bf16x16 b, f32x8 c) {
  return __builtin_amdgcn_wmma_f32_16x16x32_bf16(false, a, false, b, (short)0, c,
                                                 false, false);
}

#if HAS_ASYNC_LDS
// 16-byte async copy global -> LDS (tracked by ASYNCcnt).
// Builtin signature (per clang-22 diagnostic): (v4i AS1*, v4i AS3*, Imm, Imm).
__device__ __forceinline__ void async_cp16(const bf16* g, bf16* l) {
  __builtin_amdgcn_global_load_async_to_lds_b128(
      (__attribute__((address_space(1))) i32x4*)(uintptr_t)g,
      (__attribute__((address_space(3))) i32x4*)(unsigned)(uintptr_t)l, 0, 0);
}
#endif

// A-fragment (16x32 bf16, MxK): lane holds row M = lane&15.
// elems i -> K = (lane>=16 ? 16 : 0) + i  (16 contiguous K values).
__device__ __forceinline__ bf16x16 load_a_frag(const bf16* rowk0, int lane) {
  int lo = (lane & 16) ? 16 : 0;
  bf16x8 l = *(const bf16x8*)(rowk0 + lo);
  bf16x8 h = *(const bf16x8*)(rowk0 + 8 + lo);
  bf16x16 r;
#pragma unroll
  for (int i = 0; i < 8; ++i) { r[i] = l[i]; r[i + 8] = h[i]; }
  return r;
}

// B-fragment (32x16 bf16, KxN): lane holds column N = lane&15,
// elems i -> K = (lane>=16 ? 16 : 0) + i  (16 contiguous K values).
__device__ __forceinline__ bf16x16 load_b_frag(const bf16* colk0, int lane) {
  return *(const bf16x16*)(colk0 + ((lane & 16) ? 16 : 0));
}

// ---------------------------------------------------------------------------
// f32 -> bf16 convert
// ---------------------------------------------------------------------------
__global__ void cvt_bf16_kernel(const float* __restrict__ in,
                                bf16* __restrict__ out, int n) {
  int i = blockIdx.x * blockDim.x + threadIdx.x;
  if (i < n) out[i] = (bf16)in[i];
}

// ---------------------------------------------------------------------------
// C[M,N] = A[M,K] * B[N,K]^T.  bf16 in, f32 accum; OUT_BF16 selects store type.
// Block: 128 threads (4 waves); block tile 128x128; wave tile 64x64.
// A/B K-slabs (128x32) double-buffered in LDS via async-to-LDS copies.
// LDS rows padded to 40 bf16 (20 words) so fragment loads are conflict-free.
// ---------------------------------------------------------------------------
#define LDSW 40

template <bool OUT_BF16>
__global__ __launch_bounds__(128) void gemm_bf16_kernel(
    const bf16* __restrict__ A, const bf16* __restrict__ B,
    void* __restrict__ C, int M, int N, int K) {
  __shared__ __align__(16) bf16 Abuf[2][128 * LDSW];
  __shared__ __align__(16) bf16 Bbuf[2][128 * LDSW];

  const int t = threadIdx.x;
  const int lane = t & 31;
  const int wave = t >> 5;
  const int wm = (wave >> 1) * 64;         // local wave tile origin
  const int wn = (wave & 1) * 64;
  const int lm = lane & 15;
  const size_t mBase = (size_t)blockIdx.y * 128;
  const size_t nBase = (size_t)blockIdx.x * 128;

  // Thread t stages row t of both 128x32 slabs (4x16B each).
  auto stage = [&](int buf, int k0) {
    const bf16* gA = A + (mBase + t) * K + k0;
    const bf16* gB = B + (nBase + t) * K + k0;
    bf16* lA = &Abuf[buf][t * LDSW];
    bf16* lB = &Bbuf[buf][t * LDSW];
#if HAS_ASYNC_LDS
#pragma unroll
    for (int c = 0; c < 4; ++c) async_cp16(gA + c * 8, lA + c * 8);
#pragma unroll
    for (int c = 0; c < 4; ++c) async_cp16(gB + c * 8, lB + c * 8);
#else
#pragma unroll
    for (int c = 0; c < 4; ++c) *(bf16x8*)(lA + c * 8) = *(const bf16x8*)(gA + c * 8);
#pragma unroll
    for (int c = 0; c < 4; ++c) *(bf16x8*)(lB + c * 8) = *(const bf16x8*)(gB + c * 8);
#endif
  };

  f32x8 acc[4][4] = {};

  stage(0, 0);
  for (int k0 = 0; k0 < K; k0 += 32) {
    const int cur = (k0 >> 5) & 1;
    if (k0 + 32 < K) {
      stage(cur ^ 1, k0 + 32);
#if HAS_ASYNC_LDS
      __builtin_amdgcn_s_wait_asynccnt(8);   // current slab's 8 copies complete
#endif
    } else {
#if HAS_ASYNC_LDS
      __builtin_amdgcn_s_wait_asynccnt(0);
#endif
    }
    __syncthreads();                         // slab visible to all waves

    bf16x16 a[4], b[4];
#pragma unroll
    for (int mi = 0; mi < 4; ++mi)
      a[mi] = load_a_frag(&Abuf[cur][(wm + mi * 16 + lm) * LDSW], lane);
#pragma unroll
    for (int ni = 0; ni < 4; ++ni)
      b[ni] = load_b_frag(&Bbuf[cur][(wn + ni * 16 + lm) * LDSW], lane);
#pragma unroll
    for (int mi = 0; mi < 4; ++mi)
#pragma unroll
      for (int ni = 0; ni < 4; ++ni)
        acc[mi][ni] = wmma_bf16(a[mi], b[ni], acc[mi][ni]);

    __syncthreads();                         // done reading before restage
  }

  // C/D layout: VGPR v holds row v + 8*(lane>=16), col = lane&15.
  const int rlo = (lane & 16) ? 8 : 0;
#pragma unroll
  for (int mi = 0; mi < 4; ++mi)
#pragma unroll
    for (int ni = 0; ni < 4; ++ni)
#pragma unroll
      for (int v = 0; v < 8; ++v) {
        size_t r = mBase + mi * 16 + rlo + v;
        size_t c = nBase + ni * 16 + lm;
        if (OUT_BF16)
          ((bf16*)C)[r * N + c] = (bf16)acc[mi][ni][v];
        else
          ((float*)C)[r * N + c] = acc[mi][ni][v];
      }
}

// ---------------------------------------------------------------------------
// RoPE + relayout [s][heads*128] -> [h][s][128], optional 1/sqrt(HDIM) scale.
// grid (SEQ, heads), block 128.
// ---------------------------------------------------------------------------
__global__ void rope_relayout_kernel(const bf16* __restrict__ in,
                                     bf16* __restrict__ out,
                                     const int* __restrict__ pos,
                                     int in_cols, float scale) {
  int s = blockIdx.x, h = blockIdx.y, d = threadIdx.x;
  const bf16* row = in + (size_t)s * in_cols + h * HDIM;
  float p = (float)pos[s];
  int j = d & 63;
  // inv_freq = 10000^(-j/64) ; ln(10000)/64 = 0.14391156
  float ang = p * __expf(-(float)j * 0.14391156f);
  float x  = (float)row[d];
  float xr = (d < 64) ? -(float)row[d + 64] : (float)row[d - 64];
  out[((size_t)h * SEQ + s) * HDIM + d] =
      (bf16)((x * __cosf(ang) + xr * __sinf(ang)) * scale);
}

// V relayout-transpose: [s][NKV*128] -> [h][d][s], LDS-tiled so both global
// sides are coalesced. grid (SEQ/16, HDIM/16, NKV), block (16,16).
__global__ void v_transpose_kernel(const bf16* __restrict__ in,
                                   bf16* __restrict__ out) {
  __shared__ bf16 tile[16][17];
  int s0 = blockIdx.x * 16, d0 = blockIdx.y * 16, h = blockIdx.z;
  int tx = threadIdx.x, ty = threadIdx.y;
  tile[ty][tx] = in[(size_t)(s0 + ty) * (NKV * HDIM) + h * HDIM + d0 + tx];
  __syncthreads();
  out[((size_t)h * HDIM + d0 + ty) * SEQ + s0 + tx] = tile[tx][ty];
}

// ---------------------------------------------------------------------------
// Flash attention: one wave per (head, 16-query tile).
// Q [h][s][128] (pre-scaled by 1/sqrt(128)), K [h][s][128], Vt [h][d][s].
// Output bf16 [s][NHEADS*128].  grid (SEQ/16, NHEADS), block 32.
// ---------------------------------------------------------------------------
__global__ __launch_bounds__(32) void attn_kernel(
    const bf16* __restrict__ Q, const bf16* __restrict__ K,
    const bf16* __restrict__ Vt, bf16* __restrict__ Aout) {
  const int lane = threadIdx.x;
  const int q0 = blockIdx.x * 16;
  const int h = blockIdx.y;
  const int kvh = h / NREP;
  const bf16* Qh = Q + (size_t)h * SEQ * HDIM;
  const bf16* Kh = K + (size_t)kvh * SEQ * HDIM;
  const bf16* Vh = Vt + (size_t)kvh * HDIM * SEQ;

  __shared__ __align__(16) bf16 pls[16 * 32];  // P tile, row-major 16x32

  const int lm = lane & 15;
  const int rlo = (lane & 16) ? 8 : 0;

  // Q A-fragments for 16 rows x 128 dims (4 K-chunks of 32)
  bf16x16 qa[4];
#pragma unroll
  for (int kk = 0; kk < 4; ++kk)
    qa[kk] = load_a_frag(Qh + (size_t)(q0 + lm) * HDIM + kk * 32, lane);

  f32x8 acc[8] = {};
  float m[8], l[8];
#pragma unroll
  for (int v = 0; v < 8; ++v) { m[v] = -3.0e38f; l[v] = 0.0f; }

  const int kend = q0 + 16;  // causal: keys <= q0+15
  for (int kt = 0; kt < kend; kt += 32) {
    // scores for 32 keys: two 16x16 C tiles
    f32x8 s0 = {}, s1 = {};
#pragma unroll
    for (int kk = 0; kk < 4; ++kk) {
      bf16x16 b0 = load_b_frag(Kh + (size_t)(kt + lm) * HDIM + kk * 32, lane);
      bf16x16 b1 = load_b_frag(Kh + (size_t)(kt + 16 + lm) * HDIM + kk * 32, lane);
      s0 = wmma_bf16(qa[kk], b0, s0);
      s1 = wmma_bf16(qa[kk], b1, s1);
    }

    // online softmax (per row; rows spread over the 16-lane halves)
    float sc[8], p0[8], p1[8];
#pragma unroll
    for (int v = 0; v < 8; ++v) {
      int r = q0 + rlo + v;
      float x0 = (kt + lm <= r) ? s0[v] : -3.0e38f;
      float x1 = (kt + 16 + lm <= r) ? s1[v] : -3.0e38f;
      float mx = fmaxf(x0, x1);
#pragma unroll
      for (int off = 1; off < 16; off <<= 1) mx = fmaxf(mx, __shfl_xor(mx, off, 32));
      float nm = fmaxf(m[v], mx);
      sc[v] = __expf(m[v] - nm);
      m[v] = nm;
      p0[v] = __expf(x0 - nm);
      p1[v] = __expf(x1 - nm);
      float sum = p0[v] + p1[v];
#pragma unroll
      for (int off = 1; off < 16; off <<= 1) sum += __shfl_xor(sum, off, 32);
      l[v] = l[v] * sc[v] + sum;
    }
#pragma unroll
    for (int ct = 0; ct < 8; ++ct)
#pragma unroll
      for (int v = 0; v < 8; ++v) acc[ct][v] *= sc[v];

    // C-layout -> LDS (row-major 16x32) -> A-fragment
    __syncthreads();
#pragma unroll
    for (int v = 0; v < 8; ++v) {
      int r = rlo + v;
      pls[r * 32 + lm] = (bf16)p0[v];
      pls[r * 32 + 16 + lm] = (bf16)p1[v];
    }
    __syncthreads();
    bf16x16 pf = load_a_frag(&pls[lm * 32], lane);

    // acc += P(16x32) @ V(32x128), 8 column tiles; Vt gives contiguous K loads
#pragma unroll
    for (int ct = 0; ct < 8; ++ct) {
      bf16x16 vb = *(const bf16x16*)(Vh + (size_t)(ct * 16 + lm) * SEQ + kt +
                                     ((lane & 16) ? 16 : 0));
      acc[ct] = wmma_bf16(pf, vb, acc[ct]);
    }
  }

  // normalize and store to [s][NHEADS*128]
  float il[8];
#pragma unroll
  for (int v = 0; v < 8; ++v) il[v] = 1.0f / l[v];
#pragma unroll
  for (int ct = 0; ct < 8; ++ct)
#pragma unroll
    for (int v = 0; v < 8; ++v) {
      size_t r = (size_t)(q0 + rlo + v);
      size_t c = (size_t)(h * HDIM + ct * 16 + lm);
      Aout[r * (NHEADS * HDIM) + c] = (bf16)(acc[ct][v] * il[v]);
    }
}

// ---------------------------------------------------------------------------
extern "C" void kernel_launch(void* const* d_in, const int* in_sizes, int n_in,
                              void* d_out, int out_size, void* d_ws, size_t ws_size,
                              hipStream_t stream) {
  const float* hs = (const float*)d_in[0];
  const float* Wq = (const float*)d_in[1];
  const float* Wk = (const float*)d_in[2];
  const float* Wv = (const float*)d_in[3];
  const float* Wo = (const float*)d_in[4];
  const int* pos = (const int*)d_in[5];

  char* ws = (char*)d_ws;
  const size_t MB = 1024 * 1024;
  bf16* hsb  = (bf16*)(ws + 0 * MB);      // [4096][4096]      32 MB
  bf16* wqb  = (bf16*)(ws + 32 * MB);     // [4096][4096]      32 MB
  bf16* wkb  = (bf16*)(ws + 64 * MB);     // [1024][4096]       8 MB
  bf16* wvb  = (bf16*)(ws + 72 * MB);     // [1024][4096]       8 MB
  bf16* wob  = (bf16*)(ws + 80 * MB);     // [4096][4096]      32 MB
  bf16* qtmp = (bf16*)(ws + 112 * MB);    // [4096][4096]      32 MB (reused as attn out)
  bf16* ktmp = (bf16*)(ws + 144 * MB);    // [4096][1024]       8 MB
  bf16* vtmp = (bf16*)(ws + 152 * MB);    // [4096][1024]       8 MB
  bf16* Qr   = (bf16*)(ws + 160 * MB);    // [32][4096][128]   32 MB
  bf16* Kr   = (bf16*)(ws + 192 * MB);    // [8][4096][128]     8 MB
  bf16* Vtr  = (bf16*)(ws + 200 * MB);    // [8][128][4096]     8 MB
  bf16* attn = qtmp;                      // reuse after RoPE consumed qtmp

  // 1) converts to bf16
  const int n_hid = SEQ * HIDDEN;
  const int n_kv = NKV * HDIM * HIDDEN;
  cvt_bf16_kernel<<<(n_hid + 255) / 256, 256, 0, stream>>>(hs, hsb, n_hid);
  cvt_bf16_kernel<<<(n_hid + 255) / 256, 256, 0, stream>>>(Wq, wqb, n_hid);
  cvt_bf16_kernel<<<(n_kv + 255) / 256, 256, 0, stream>>>(Wk, wkb, n_kv);
  cvt_bf16_kernel<<<(n_kv + 255) / 256, 256, 0, stream>>>(Wv, wvb, n_kv);
  cvt_bf16_kernel<<<(n_hid + 255) / 256, 256, 0, stream>>>(Wo, wob, n_hid);

  // 2) QKV projections
  gemm_bf16_kernel<true><<<dim3(HIDDEN / 128, SEQ / 128), 128, 0, stream>>>(
      hsb, wqb, qtmp, SEQ, HIDDEN, HIDDEN);
  gemm_bf16_kernel<true><<<dim3((NKV * HDIM) / 128, SEQ / 128), 128, 0, stream>>>(
      hsb, wkb, ktmp, SEQ, NKV * HDIM, HIDDEN);
  gemm_bf16_kernel<true><<<dim3((NKV * HDIM) / 128, SEQ / 128), 128, 0, stream>>>(
      hsb, wvb, vtmp, SEQ, NKV * HDIM, HIDDEN);

  // 3) RoPE + relayout (fold 1/sqrt(128) into Q); V transpose
  rope_relayout_kernel<<<dim3(SEQ, NHEADS), HDIM, 0, stream>>>(
      qtmp, Qr, pos, NHEADS * HDIM, 0.08838834764831845f);
  rope_relayout_kernel<<<dim3(SEQ, NKV), HDIM, 0, stream>>>(
      ktmp, Kr, pos, NKV * HDIM, 1.0f);
  v_transpose_kernel<<<dim3(SEQ / 16, HDIM / 16, NKV), dim3(16, 16), 0, stream>>>(
      vtmp, Vtr);

  // 4) causal flash attention (one wave per 16-query tile per head)
  attn_kernel<<<dim3(SEQ / 16, NHEADS), 32, 0, stream>>>(Qr, Kr, Vtr, attn);

  // 5) output projection -> f32 d_out
  gemm_bf16_kernel<false><<<dim3(HIDDEN / 128, SEQ / 128), 128, 0, stream>>>(
      attn, wob, d_out, SEQ, HIDDEN, HIDDEN);
}
